// MCLSTM_65386582114604
// MI455X (gfx1250) — matrix operand order repaired and checked by reading, not compiled, not run on hardware
//
#include <hip/hip_runtime.h>
#include <stdint.h>

typedef _Float16 f16;
typedef _Float16 v16h __attribute__((ext_vector_type(16)));
typedef float    v8f  __attribute__((ext_vector_type(8)));

#define B_   1024
#define T_   365
#define TM1  364
#define CC   24
#define INN  29
#define HEAD 49
#define TOT  625
#define KP   32      // padded K (29 -> 32)
#define NP   640     // padded N (625 -> 640, 40 tiles of 16)
#define GS   644     // scan g LDS row stride (floats): 644%64=4 -> no M/M+8 bank clash
#define GS2  580     // post g LDS row stride (576 cols)
#define XSTR 36      // x-row stride (state 0..23, aux 24..28, zero pad 29..31); gcd(36,64)=4 -> conflict-free

// ---- LDS budgets (bytes) ----
// scan: g(16*644*4=41216) + x(16*36*4=2304) + ct/mx/den(3*1536) + cpot/rmax/rsum/assim(4*64)
//       + parA(96) + parG(32) = 48512
#define LDS_SCAN 48512
// post: g(16*580*4=37120) + x(2304) + ct/mx/den(3*1536) = 44032
#define LDS_POST 44032

__device__ __forceinline__ float sigmoidf_(float x) { return 1.0f / (1.0f + __expf(-x)); }

__device__ __forceinline__ void compute_aux(const float* __restrict__ X,
                                            const float* __restrict__ ORY,
                                            const float* __restrict__ Ncum,
                                            int b, int t, float* aux5, float* cpot) {
  const float4 x4 = ((const float4*)X)[b * T_ + t];     // X[b][t][0..3], 16B aligned
  float dvs   = ORY[((size_t)b * T_ + t) * 7];
  float t_ave = 0.5f * (x4.y + x4.z);
  float tv    = fminf(fmaxf(t_ave * 50.0f, 10.0f), 40.0f);
  float eff   = 0.54f - (tv - 10.0f) * (1.0f / 30.0f) * (0.54f - 0.36f);
  float R2C   = 2.0f * 0.5f * eff * (1.0f / 3.6f) * (12.0f / 44.0f);
  aux5[0] = dvs; aux5[1] = x4.x; aux5[2] = x4.y; aux5[3] = x4.z;
  aux5[4] = Ncum[b * T_ + t];
  *cpot = x4.x * R2C;
}

// build A fragment (16x32 f16, wave32 layout) from padded x-rows in LDS:
// element e <-> K = hf*16 + e; 4x ds_load_b128 + packed converts
__device__ __forceinline__ v16h build_afrag(const float* sX, int m_a, int hf) {
  const float* xr = sX + m_a * XSTR;
  float4 a0 = *(const float4*)(xr + (hf << 4));
  float4 a1 = *(const float4*)(xr + (hf << 4) + 4);
  float4 a2 = *(const float4*)(xr + (hf << 4) + 8);
  float4 a3 = *(const float4*)(xr + (hf << 4) + 12);
  v16h a;
  a[0]  = (f16)a0.x; a[1]  = (f16)a0.y; a[2]  = (f16)a0.z; a[3]  = (f16)a0.w;
  a[4]  = (f16)a1.x; a[5]  = (f16)a1.y; a[6]  = (f16)a1.z; a[7]  = (f16)a1.w;
  a[8]  = (f16)a2.x; a[9]  = (f16)a2.y; a[10] = (f16)a2.z; a[11] = (f16)a2.w;
  a[12] = (f16)a3.x; a[13] = (f16)a3.y; a[14] = (f16)a3.z; a[15] = (f16)a3.w;
  return a;
}

// ---------- prep: cumsum(N) per row + all_day[:,0,:] = ORY[:,0,:] ----------
__global__ void k_prep_misc(const float* __restrict__ X, const float* __restrict__ ORY,
                            float* __restrict__ Ncum, float* __restrict__ out_allday) {
  int b = blockIdx.x * blockDim.x + threadIdx.x;
  if (b >= B_) return;
  float acc = 0.f;
  for (int t = 0; t < T_; ++t) {
    acc += X[((size_t)b * T_ + t) * 4 + 3];
    Ncum[b * T_ + t] = acc;
  }
  for (int i = 0; i < 7; ++i)
    out_allday[(size_t)b * T_ * 7 + i] = ORY[(size_t)b * T_ * 7 + i];
}

// ---------- prep: Wm = W*prior, f16, padded, column-major (K contiguous) ----------
__global__ void k_prep_w(const float* __restrict__ W, f16* __restrict__ Wt) {
  int idx = blockIdx.x * blockDim.x + threadIdx.x;   // idx = n*32 + k
  if (idx >= NP * KP) return;
  int n = idx >> 5, k = idx & 31;
  float v = 0.f;
  if (k < INN && n < TOT) {
    v = W[k * TOT + n];
    if (k >= CC + 1 && n >= HEAD) v = 0.f;           // prior mask
  }
  Wt[idx] = (f16)v;
}

// ---------- the 364-step recurrent scan: 16 batch rows per block ----------
__global__ void __launch_bounds__(256, 1) k_scan(
    const float* __restrict__ X, const float* __restrict__ ORY,
    const float* __restrict__ bias, const float* __restrict__ parA,
    const float* __restrict__ parG, const f16* __restrict__ Wt,
    const float* __restrict__ Ncum,
    float* __restrict__ out_allday, float* __restrict__ out_ccell) {
  extern __shared__ ulong4 smem_raw[];
  char* p = (char*)smem_raw;
  float* sG     = (float*)p;    p += 16 * GS * 4;
  float* sX     = (float*)p;    p += 16 * XSTR * 4;   // state | aux | zero pad
  float* sCt    = (float*)p;    p += 16 * CC * 4;
  float* sMx    = (float*)p;    p += 16 * CC * 4;
  float* sDen   = (float*)p;    p += 16 * CC * 4;
  float* sCpot  = (float*)p;    p += 16 * 4;
  float* sRMax  = (float*)p;    p += 16 * 4;
  float* sRSum  = (float*)p;    p += 16 * 4;
  float* sAssim = (float*)p;    p += 16 * 4;
  float* sParA  = (float*)p;    p += CC * 4;
  float* sParG  = (float*)p;    p += 8 * 4;

  const int tid  = threadIdx.x;
  const int lane = tid & 31;
  const int wv   = tid >> 5;
  const int b0   = blockIdx.x * 16;
  const int m_a  = lane & 15;
  const int hf   = lane >> 4;
  const int kb   = hf << 4;

  // one-time: zero x rows (incl. pad), load small params
  for (int i = tid; i < 16 * XSTR; i += 256) sX[i] = 0.f;
  if (tid < CC) sParA[tid] = parA[tid];
  if (tid < 8)  sParG[tid] = parG[tid];

  // hoist B-fragments + bias for this wave's 5 N-tiles into registers (invariant over t)
  v16h bfr[5]; float bvr[5];
  #pragma unroll
  for (int i = 0; i < 5; ++i) {
    const int nn = (wv * 5 + i) * 16 + m_a;
    bfr[i] = *(const v16h*)(Wt + nn * KP + kb);        // 32B global load, K-major
    bvr[i] = (nn < TOT) ? bias[nn] : 0.0f;
  }
  __syncthreads();

  for (int t = 0; t < TM1; ++t) {
    // P0: per-row aux + c_pot -> padded x row
    if (tid < 16) {
      float a5[5], cp;
      compute_aux(X, ORY, Ncum, b0 + tid, t, a5, &cp);
      #pragma unroll
      for (int i = 0; i < 5; ++i) sX[tid * XSTR + CC + i] = a5[i];
      sCpot[tid] = cp;
    }
    __syncthreads();

    // GEMM: g[16x625] = x[16x29] @ Wm + b ; B/bias in registers -> wmma back-to-back
    v16h afrag = build_afrag(sX, m_a, hf);
    #pragma unroll
    for (int i = 0; i < 5; ++i) {
      v8f acc;
      #pragma unroll
      for (int r = 0; r < 8; ++r) acc[r] = bvr[i];     // bias folded into accumulator
      acc = __builtin_amdgcn_wmma_f32_16x16x32_f16(false, afrag, false, bfr[i],
                                                   (short)0, acc, false, false);
      const int nn = (wv * 5 + i) * 16 + m_a;
      #pragma unroll
      for (int r = 0; r < 8; ++r)                      // C layout: M = r + 8*hf, N = lane%16
        sG[((hf << 3) + r) * GS + nn] = acc[r];
    }
    __syncthreads();

    // P1: row stats (assim, part-softmax) + red-softmax per-c stats
    if (tid < 16) {
      const float* g = sG + tid * GS;
      float assim = sigmoidf_(g[0]);
      float mx = -3.0e38f;
      for (int c = 0; c < CC; ++c) mx = fmaxf(mx, g[1 + c]);
      float sm = 0.f;
      for (int c = 0; c < CC; ++c) sm += __expf(g[1 + c] - mx);
      sRMax[tid] = mx; sRSum[tid] = sm;
      sAssim[tid] = assim * sCpot[tid];                // fold C_in = assim*c_pot
    }
    for (int idx = tid; idx < 16 * CC; idx += 256) {
      int m = idx / CC, c = idx - (idx / CC) * CC;
      const float* g = sG + m * GS + HEAD + c * CC;
      float mx = -3.0e38f;
      for (int d = 0; d < CC; ++d) mx = fmaxf(mx, g[d]);
      float sm = 0.f;
      for (int d = 0; d < CC; ++d) sm += __expf(g[d] - mx);
      sMx[idx] = mx; sDen[idx] = sm;
    }
    __syncthreads();

    // P2: gated state, pre-divided by its red denominator
    for (int idx = tid; idx < 16 * CC; idx += 256) {
      int m = idx / CC, c = idx - m * CC;
      float part = __expf(sG[m * GS + 1 + c] - sRMax[m]) / sRSum[m];
      float cons = sigmoidf_(sG[m * GS + 1 + CC + c]);
      float ct = (sX[m * XSTR + c] + sAssim[m] * part) * (1.f - cons);
      sCt[idx] = ct / sDen[idx];
    }
    __syncthreads();

    // P3: C_new[d] = sum_c ct[c]/den[c] * exp(g_red[c][d]-mx[c])
    for (int idx = tid; idx < 16 * CC; idx += 256) {
      int m = idx / CC, d = idx - m * CC;
      const float* g = sG + m * GS + HEAD + d;
      float s = 0.f;
      #pragma unroll
      for (int c = 0; c < CC; ++c)
        s += sCt[m * CC + c] * __expf(g[c * CC] - sMx[m * CC + c]);
      sX[m * XSTR + d] = s;
      out_ccell[(((size_t)(b0 + m)) * TM1 + t) * CC + d] = s;
    }
    __syncthreads();

    // P4: daily scalar outputs -> all_day[:, t+1, :]
    if (tid < 16) {
      const float* st = sX + tid * XSTR;
      float pai = 0.f, lea = 0.f, ste = 0.f, gra = 0.f, yie = 0.f;
      for (int c = 0; c < CC; ++c) pai += fabsf(st[c] * sParA[c]);
      for (int c = 0; c < 8; ++c)  lea += st[c];
      for (int c = 8; c < 16; ++c) ste += st[c];
      for (int c = 16; c < 24; ++c) gra += st[c];
      for (int c = 0; c < 8; ++c)  yie += fabsf(st[16 + c] * sParG[c]);
      lea *= (1.0f / 0.419f); ste *= (1.0f / 0.431f);
      gra *= (1.0f / 0.487f); yie *= (1.0f / 0.487f);
      float agb = lea + ste + gra;
      int b = b0 + tid;
      float dvs1 = ORY[((size_t)b * T_ + (t + 1)) * 7];
      float* o = out_allday + ((size_t)b * T_ + (t + 1)) * 7;
      o[0] = dvs1; o[1] = pai; o[2] = lea; o[3] = ste; o[4] = gra; o[5] = agb; o[6] = yie;
    }
    __syncthreads();
  }
}

// ---------- parallel post-pass: g2 = Xc @ Wm[:,49:] + b[49:], softmax, matvec ----------
__global__ void __launch_bounds__(256, 1) k_post(
    const float* __restrict__ X, const float* __restrict__ ORY,
    const float* __restrict__ bias, const f16* __restrict__ Wt,
    const float* __restrict__ Ncum, const float* __restrict__ ccell,
    float* __restrict__ out_cconv) {
  extern __shared__ ulong4 smem_raw2[];
  char* p = (char*)smem_raw2;
  float* sG  = (float*)p; p += 16 * GS2 * 4;
  float* sX  = (float*)p; p += 16 * XSTR * 4;
  float* sCt = (float*)p; p += 16 * CC * 4;
  float* sMx = (float*)p; p += 16 * CC * 4;
  float* sDen= (float*)p; p += 16 * CC * 4;

  const int tid  = threadIdx.x;
  const int lane = tid & 31;
  const int wv   = tid >> 5;
  const int m_a  = lane & 15;
  const int hf   = lane >> 4;
  const int kb   = hf << 4;
  const int ntiles = (B_ * TM1) / 16;                  // 23296 M-tiles

  for (int i = tid; i < 16 * XSTR; i += 256) sX[i] = 0.f;   // zero pad cols once

  // hoist B-fragments + bias (invariant across all M-tiles): jt = wv, wv+8, ... (<36)
  v16h bfr[5]; float bvr[5];
  #pragma unroll
  for (int i = 0; i < 5; ++i) {
    int jt = wv + 8 * i; if (jt > 35) jt = 35;         // clamp (dup load, store guarded)
    const int nn = HEAD + jt * 16 + m_a;
    bfr[i] = *(const v16h*)(Wt + nn * KP + kb);
    bvr[i] = bias[nn];
  }
  __syncthreads();

  for (int tile = blockIdx.x; tile < ntiles; tile += gridDim.x) {
    const int r0 = tile * 16;
    for (int idx = tid; idx < 16 * CC; idx += 256) {   // rows of ccell are contiguous
      int m = idx / CC, c = idx - m * CC;
      sX[m * XSTR + c] = ccell[(size_t)r0 * CC + idx];
    }
    if (tid < 16) {
      int r = r0 + tid, b = r / TM1, t = r - b * TM1;
      float a5[5], cp;
      compute_aux(X, ORY, Ncum, b, t, a5, &cp);
      #pragma unroll
      for (int i = 0; i < 5; ++i) sX[tid * XSTR + CC + i] = a5[i];
    }
    __syncthreads();

    v16h afrag = build_afrag(sX, m_a, hf);
    #pragma unroll
    for (int i = 0; i < 5; ++i) {
      const int jt = wv + 8 * i;                       // wave-uniform validity
      if (jt < 36) {
        v8f acc;
        #pragma unroll
        for (int r = 0; r < 8; ++r) acc[r] = bvr[i];
        acc = __builtin_amdgcn_wmma_f32_16x16x32_f16(false, afrag, false, bfr[i],
                                                     (short)0, acc, false, false);
        const int ng = jt * 16 + m_a;
        #pragma unroll
        for (int r = 0; r < 8; ++r)
          sG[((hf << 3) + r) * GS2 + ng] = acc[r];
      }
    }
    __syncthreads();

    for (int idx = tid; idx < 16 * CC; idx += 256) {
      int m = idx / CC, c = idx - m * CC;
      const float* g = sG + m * GS2 + c * CC;
      float mx = -3.0e38f;
      for (int d = 0; d < CC; ++d) mx = fmaxf(mx, g[d]);
      float sm = 0.f;
      for (int d = 0; d < CC; ++d) sm += __expf(g[d] - mx);
      sMx[idx] = mx; sDen[idx] = sm;
    }
    __syncthreads();
    for (int idx = tid; idx < 16 * CC; idx += 256) {
      int m = idx / CC, c = idx - m * CC;
      sCt[idx] = sX[m * XSTR + c] / sDen[idx];
    }
    __syncthreads();
    for (int idx = tid; idx < 16 * CC; idx += 256) {
      int m = idx / CC, d = idx - m * CC;
      const float* g = sG + m * GS2 + d;
      float s = 0.f;
      #pragma unroll
      for (int c = 0; c < CC; ++c)
        s += sCt[m * CC + c] * __expf(g[c * CC] - sMx[m * CC + c]);
      out_cconv[(size_t)(r0 + m) * CC + d] = s;
    }
    __syncthreads();
  }
}

extern "C" void kernel_launch(void* const* d_in, const int* in_sizes, int n_in,
                              void* d_out, int out_size, void* d_ws, size_t ws_size,
                              hipStream_t stream) {
  const float* X    = (const float*)d_in[0];
  const float* ORY  = (const float*)d_in[1];
  const float* W    = (const float*)d_in[2];
  const float* bb   = (const float*)d_in[3];
  const float* parA = (const float*)d_in[4];
  const float* parG = (const float*)d_in[5];

  float* out_allday = (float*)d_out;                         // B*T*7
  float* out_ccell  = out_allday + (size_t)B_ * T_ * 7;      // B*364*24
  float* out_cconv  = out_ccell + (size_t)B_ * TM1 * CC;     // B*364*24

  f16*   Wt   = (f16*)d_ws;                                  // 40 KB
  float* Ncum = (float*)((char*)d_ws + 64 * 1024);           // B*T floats (~1.5 MB)

  k_prep_misc<<<(B_ + 255) / 256, 256, 0, stream>>>(X, ORY, Ncum, out_allday);
  k_prep_w<<<(NP * KP + 255) / 256, 256, 0, stream>>>(W, Wt);

  hipFuncSetAttribute((const void*)k_scan, hipFuncAttributeMaxDynamicSharedMemorySize, LDS_SCAN);
  hipFuncSetAttribute((const void*)k_post, hipFuncAttributeMaxDynamicSharedMemorySize, LDS_POST);

  k_scan<<<B_ / 16, 256, LDS_SCAN, stream>>>(X, ORY, bb, parA, parG, Wt, Ncum,
                                             out_allday, out_ccell);
  k_post<<<1456, 256, LDS_POST, stream>>>(X, ORY, bb, Wt, Ncum, out_ccell, out_cconv);
}
